// _OpenClipAttentionAdapter_63995012710884
// MI455X (gfx1250) — matrix operand order repaired and checked by reading, not compiled, not run on hardware
//
#include <hip/hip_runtime.h>
#include <hip/hip_bf16.h>

typedef __attribute__((ext_vector_type(16))) __bf16 bf16x16;
typedef __attribute__((ext_vector_type(8)))  __bf16 bf16x8;
typedef __attribute__((ext_vector_type(8)))  float  f32x8;

static __device__ __forceinline__ f32x8 wmma_bf16(bf16x16 a, bf16x16 b, f32x8 c) {
    // D = A(16x32 bf16) * B(32x16 bf16) + C(16x16 f32)
    return __builtin_amdgcn_wmma_f32_16x16x32_bf16(false, a, false, b, (short)0, c, false, false);
}

static __device__ __forceinline__ bf16x16 cat8(bf16x8 lo, bf16x8 hi) {
    return __builtin_shufflevector(lo, hi, 0,1,2,3,4,5,6,7,8,9,10,11,12,13,14,15);
}

// Low 32 bits of a generic pointer to __shared__ = LDS byte offset.
static __device__ __forceinline__ unsigned lds_off(const void* p) {
    return (unsigned)(unsigned long long)p;
}

// CDNA5 async global->LDS copy (16B per lane), tracked by ASYNCcnt.
static __device__ __forceinline__ void async_b128(unsigned ldsoff, unsigned goff, const void* base) {
    asm volatile("global_load_async_to_lds_b128 %0, %1, %2"
                 :: "v"(ldsoff), "v"(goff), "s"(base) : "memory");
}

template <int N>
static __device__ __forceinline__ void wait_asynccnt() {
    asm volatile("s_wait_asynccnt %0" :: "n"(N) : "memory");
}

// ---------------------------------------------------------------------------
// f32 -> bf16 conversion
// ---------------------------------------------------------------------------
__global__ void cvt_bf16(const float* __restrict__ src, __bf16* __restrict__ dst, int n) {
    int i = blockIdx.x * blockDim.x + threadIdx.x;
    int stride = gridDim.x * blockDim.x;
    for (; i < n; i += stride) dst[i] = (__bf16)src[i];
}

// ---------------------------------------------------------------------------
// Y[m][n] = sum_k X[m][k] * W[n][k] + bias[n]
// X: MxK bf16 row-major, W: NxK bf16 row-major (x @ W^T), bias f32.
// Block tile 128x256, K-step 32, double-buffered async LDS staging.
// 8 waves: 2 (M) x 4 (N); wave tile 64x64 -> 16 WMMA accs per wave.
// ---------------------------------------------------------------------------
template <bool OUTF32>
__global__ __launch_bounds__(256) void gemm_bias(const __bf16* __restrict__ X,
                                                 const __bf16* __restrict__ W,
                                                 const float* __restrict__ bias,
                                                 void* __restrict__ Yv,
                                                 int M, int N, int K) {
    constexpr int BM = 128, BN = 256, BK = 32, LDT = 40;  // padded LDS stride
    __shared__ __align__(16) __bf16 As[2][BM * LDT];
    __shared__ __align__(16) __bf16 Ws[2][BN * LDT];

    const int tid  = threadIdx.x;
    const int lane = tid & 31;
    const int wid  = tid >> 5;
    const int wm0  = (wid >> 2) * 64;   // wave M origin in block tile
    const int wn0  = (wid & 3)  * 64;   // wave N origin in block tile
    const int m0   = blockIdx.y * BM;
    const int n0   = blockIdx.x * BN;

    const int ldrA = tid >> 1;          // A staging row 0..127 (16 elems each)
    const int ldcA = (tid & 1) * 16;
    const int ldrW = tid;               // W staging row 0..255 (32 elems each)

    const int offA = (lane >> 4) * 8;   // A-frag K split per lane half
    const int offB = (lane >> 4) * 8;   // B-frag K split per lane half
    const int l15  = lane & 15;

    f32x8 acc[4][4] = {};

    auto issue = [&](int buf, int k0) {
        unsigned ga = (unsigned)(((unsigned)(m0 + ldrA) * (unsigned)K + k0 + ldcA) * 2u);
        unsigned la = lds_off(&As[buf][ldrA * LDT + ldcA]);
        async_b128(la,      ga,      X);
        async_b128(la + 16, ga + 16, X);
        unsigned gw = (unsigned)(((unsigned)(n0 + ldrW) * (unsigned)K + k0) * 2u);
        unsigned lw = lds_off(&Ws[buf][ldrW * LDT]);
#pragma unroll
        for (int j = 0; j < 4; ++j) async_b128(lw + j * 16, gw + j * 16, W);
    };

    const int KT = K / BK;
    issue(0, 0);

    for (int kt = 0; kt < KT; ++kt) {
        __syncthreads();  // prior readers of buf[(kt+1)&1] are done everywhere
        if (kt + 1 < KT) {
            issue((kt + 1) & 1, (kt + 1) * BK);
            wait_asynccnt<6>();   // tile kt resident; tile kt+1 still in flight
        } else {
            wait_asynccnt<0>();
        }
        __syncthreads();  // tile kt visible to all waves

        const __bf16* as = As[kt & 1];
        const __bf16* ws = Ws[kt & 1];

        bf16x16 afrag[4];
        bf16x16 bfrag[4];
#pragma unroll
        for (int mi = 0; mi < 4; ++mi) {
            int r = wm0 + mi * 16 + l15;
            bf16x8 lo = *(const bf16x8*)&as[r * LDT + offA];
            bf16x8 hi = *(const bf16x8*)&as[r * LDT + offA + 16];
            afrag[mi] = cat8(lo, hi);
        }
#pragma unroll
        for (int ni = 0; ni < 4; ++ni) {
            int r = wn0 + ni * 16 + l15;
            bf16x8 lo = *(const bf16x8*)&ws[r * LDT + offB];
            bf16x8 hi = *(const bf16x8*)&ws[r * LDT + offB + 16];
            bfrag[ni] = cat8(lo, hi);
        }
#pragma unroll
        for (int mi = 0; mi < 4; ++mi)
#pragma unroll
            for (int ni = 0; ni < 4; ++ni)
                acc[mi][ni] = wmma_bf16(afrag[mi], bfrag[ni], acc[mi][ni]);
    }

    // Epilogue: C layout — lane = col (N=lane%16), slot r = row (lane<16 ? r : 8+r)
#pragma unroll
    for (int ni = 0; ni < 4; ++ni) {
        int col = n0 + wn0 + ni * 16 + l15;
        float bv = bias[col];
#pragma unroll
        for (int mi = 0; mi < 4; ++mi) {
            int rbase = m0 + wm0 + mi * 16 + (lane >> 4) * 8;
#pragma unroll
            for (int r = 0; r < 8; ++r) {
                float v = acc[mi][ni][r] + bv;
                if (OUTF32)
                    ((float*)Yv)[(size_t)(rbase + r) * N + col] = v;
                else
                    ((__bf16*)Yv)[(size_t)(rbase + r) * N + col] = (__bf16)v;
            }
        }
    }
}

// ---------------------------------------------------------------------------
// Flash attention, bf16 WMMA. Q/K/V: (B*N, D) bf16 row-major; head h uses
// columns [h*64, h*64+64). One WG = (b, h, 128 q rows); 8 waves x 16 q rows.
// kv processed in chunks of 32 with online softmax. K staged via async-to-LDS.
// ---------------------------------------------------------------------------
__global__ __launch_bounds__(256) void flash_attn(const __bf16* __restrict__ Q,
                                                  const __bf16* __restrict__ Kb,
                                                  const __bf16* __restrict__ Vb,
                                                  __bf16* __restrict__ Ctx) {
    constexpr int NN = 1024, DM = 1024;
    constexpr float SCALE = 0.125f;  // 64^-0.5

    __shared__ __align__(16) __bf16 Ks[32 * 80];       // [kv row][d], padded
    __shared__ __align__(16) __bf16 Vt[64 * 40];       // [d][kv row] (transposed)
    __shared__ __align__(16) __bf16 Ps[8 * 16 * 40];   // per-wave P tile [row][kv]

    const int b  = blockIdx.z;
    const int h  = blockIdx.y;
    const int q0 = blockIdx.x * 128;
    const int tid  = threadIdx.x;
    const int lane = tid & 31;
    const int w    = tid >> 5;

    const size_t base = (size_t)b * NN * DM + (size_t)h * 64;
    const __bf16* Qp = Q  + base;
    const __bf16* Kp = Kb + base;
    const __bf16* Vp = Vb + base;

    const int offA = (lane >> 4) * 8;
    const int offB = (lane >> 4) * 8;
    const int l15  = lane & 15;

    // Load the wave's two Q A-fragments (16 rows x 64 d = 2 K-chunks of 32)
    bf16x16 qf[2];
    {
        int qrow = q0 + w * 16 + l15;
#pragma unroll
        for (int dc = 0; dc < 2; ++dc) {
            const __bf16* g = Qp + (size_t)qrow * DM + dc * 32;
            bf16x8 lo = *(const bf16x8*)(g + offA);
            bf16x8 hi = *(const bf16x8*)(g + offA + 16);
            qf[dc] = cat8(lo, hi);
        }
    }

    f32x8 o[4] = {};
    float m_[8], l_[8];
#pragma unroll
    for (int r = 0; r < 8; ++r) { m_[r] = -__builtin_inff(); l_[r] = 0.f; }

    const int sr = tid >> 3;        // staging kv row 0..31
    const int sc = (tid & 7) * 8;   // staging d col

    __bf16* pw = &Ps[w * 16 * 40];

    for (int kv0 = 0; kv0 < NN; kv0 += 32) {
        __syncthreads();
        {   // K chunk via async copy; V chunk transposed through VGPRs
            unsigned goff = (unsigned)(((unsigned)(kv0 + sr) * (unsigned)DM + sc) * 2u);
            async_b128(lds_off(&Ks[sr * 80 + sc]), goff, Kp);
            const __bf16* vg = Vp + (size_t)(kv0 + sr) * DM + sc;
            bf16x8 vv = *(const bf16x8*)vg;
#pragma unroll
            for (int i = 0; i < 8; ++i) Vt[(sc + i) * 40 + sr] = vv[i];
        }
        wait_asynccnt<0>();
        __syncthreads();

        // S = Q K^T over this 32-kv chunk: two 16x16 tiles
        f32x8 s[2] = {};
#pragma unroll
        for (int t = 0; t < 2; ++t) {
            int kcol = t * 16 + l15;
#pragma unroll
            for (int dc = 0; dc < 2; ++dc) {
                bf16x8 lo = *(const bf16x8*)&Ks[kcol * 80 + dc * 32 + offB];
                bf16x8 hi = *(const bf16x8*)&Ks[kcol * 80 + dc * 32 + offB + 16];
                s[t] = wmma_bf16(qf[dc], cat8(lo, hi), s[t]);
            }
            s[t] = s[t] * SCALE;
        }

        // Online softmax. Row r of slot lives on one 16-lane half; reduce with
        // xor masks 1/2/4/8 (stay within half in wave32).
        float p0[8], p1[8], alpha[8];
#pragma unroll
        for (int r = 0; r < 8; ++r) {
            float cm = fmaxf(s[0][r], s[1][r]);
            cm = fmaxf(cm, __shfl_xor(cm, 1));
            cm = fmaxf(cm, __shfl_xor(cm, 2));
            cm = fmaxf(cm, __shfl_xor(cm, 4));
            cm = fmaxf(cm, __shfl_xor(cm, 8));
            float mn = fmaxf(m_[r], cm);
            alpha[r] = __expf(m_[r] - mn);
            p0[r] = __expf(s[0][r] - mn);
            p1[r] = __expf(s[1][r] - mn);
            float rs = p0[r] + p1[r];
            rs += __shfl_xor(rs, 1);
            rs += __shfl_xor(rs, 2);
            rs += __shfl_xor(rs, 4);
            rs += __shfl_xor(rs, 8);
            l_[r] = l_[r] * alpha[r] + rs;
            m_[r] = mn;
        }
#pragma unroll
        for (int dt = 0; dt < 4; ++dt)
#pragma unroll
            for (int r = 0; r < 8; ++r) o[dt][r] *= alpha[r];

        // Re-layout P (C layout) -> A fragment via per-wave LDS tile
        {
            int prow = (lane >> 4) * 8;
#pragma unroll
            for (int r = 0; r < 8; ++r) {
                pw[(prow + r) * 40 + l15]      = (__bf16)p0[r];
                pw[(prow + r) * 40 + 16 + l15] = (__bf16)p1[r];
            }
        }
        __syncthreads();

        bf16x16 pf;
        {
            bf16x8 lo = *(const bf16x8*)&pw[l15 * 40 + offA];
            bf16x8 hi = *(const bf16x8*)&pw[l15 * 40 + offA + 16];
            pf = cat8(lo, hi);
        }

        // O += P V : four 16x16 d-tiles, one 32-deep WMMA each
#pragma unroll
        for (int dt = 0; dt < 4; ++dt) {
            int ncol = dt * 16 + l15;
            bf16x8 lo = *(const bf16x8*)&Vt[ncol * 40 + offB];
            bf16x8 hi = *(const bf16x8*)&Vt[ncol * 40 + offB + 16];
            o[dt] = wmma_bf16(pf, cat8(lo, hi), o[dt]);
        }
    }

    // Normalize and store context (bf16, token-major (B*N, D))
    float inv[8];
#pragma unroll
    for (int r = 0; r < 8; ++r) inv[r] = 1.f / l_[r];
#pragma unroll
    for (int dt = 0; dt < 4; ++dt) {
        int col = h * 64 + dt * 16 + l15;
#pragma unroll
        for (int r = 0; r < 8; ++r) {
            int row = b * NN + q0 + w * 16 + (lane >> 4) * 8 + r;
            Ctx[(size_t)row * DM + col] = (__bf16)(o[dt][r] * inv[r]);
        }
    }
}

// ---------------------------------------------------------------------------
// Host-side launch
// ---------------------------------------------------------------------------
extern "C" void kernel_launch(void* const* d_in, const int* in_sizes, int n_in,
                              void* d_out, int out_size, void* d_ws, size_t ws_size,
                              hipStream_t stream) {
    (void)in_sizes; (void)n_in; (void)out_size; (void)ws_size;

    const int B = 4, N = 1024, D = 1024;
    const int M = B * N;                    // 4096 tokens
    const float* hs = (const float*)d_in[0];
    const float* qw = (const float*)d_in[1];
    const float* qb = (const float*)d_in[2];
    const float* kw = (const float*)d_in[3];
    const float* kb = (const float*)d_in[4];
    const float* vw = (const float*)d_in[5];
    const float* vb = (const float*)d_in[6];
    const float* ow = (const float*)d_in[7];
    const float* ob = (const float*)d_in[8];

    char* ws = (char*)d_ws;
    const size_t MB = 1u << 20;
    __bf16* xbf  = (__bf16*)(ws + 0 * MB);   // 8 MB   (4096x1024 bf16)
    __bf16* qwbf = (__bf16*)(ws + 8 * MB);   // 2 MB
    __bf16* kwbf = (__bf16*)(ws + 10 * MB);  // 2 MB
    __bf16* vwbf = (__bf16*)(ws + 12 * MB);  // 2 MB
    __bf16* owbf = (__bf16*)(ws + 14 * MB);  // 2 MB
    __bf16* qbf  = (__bf16*)(ws + 16 * MB);  // 8 MB
    __bf16* kbf  = (__bf16*)(ws + 24 * MB);  // 8 MB
    __bf16* vbf  = (__bf16*)(ws + 32 * MB);  // 8 MB
    __bf16* ctx  = (__bf16*)(ws + 40 * MB);  // 8 MB

    cvt_bf16<<<2048, 256, 0, stream>>>(hs, xbf, M * D);
    cvt_bf16<<<1024, 256, 0, stream>>>(qw, qwbf, D * D);
    cvt_bf16<<<1024, 256, 0, stream>>>(kw, kwbf, D * D);
    cvt_bf16<<<1024, 256, 0, stream>>>(vw, vwbf, D * D);
    cvt_bf16<<<1024, 256, 0, stream>>>(ow, owbf, D * D);

    dim3 gg(D / 256, M / 128);               // (4, 32)
    gemm_bias<false><<<gg, 256, 0, stream>>>(xbf, qwbf, qb, qbf, M, D, D);
    gemm_bias<false><<<gg, 256, 0, stream>>>(xbf, kwbf, kb, kbf, M, D, D);
    gemm_bias<false><<<gg, 256, 0, stream>>>(xbf, vwbf, vb, vbf, M, D, D);

    dim3 ga(N / 128, 16, B);                 // (8, 16, 4)
    flash_attn<<<ga, 256, 0, stream>>>(qbf, kbf, vbf, ctx);

    gemm_bias<true><<<gg, 256, 0, stream>>>(ctx, owbf, ob, d_out, M, D, D);
}
